// AttentionQK_19112604467517
// MI455X (gfx1250) — matrix operand order repaired and checked by reading, not compiled, not run on hardware
//
#include <hip/hip_runtime.h>
#include <hip/hip_bf16.h>
#include <math.h>

// Problem constants (from reference)
#define BATCH 32
#define SLEN  8192
#define ENCD  512
#define QDIM  128
#define CDIM  256
#define NEGS  0.3f
#define STILES (SLEN / 16)          // 512 S-tiles per batch row

// Trivial vector types (safe inside unions, unlike HIP's uint4)
typedef __attribute__((ext_vector_type(16))) __bf16    v16bf;
typedef __attribute__((ext_vector_type(8)))  float     v8f;
typedef __attribute__((ext_vector_type(4)))  unsigned  u32x4;
typedef __attribute__((ext_vector_type(4)))  float     f32x4;
typedef __attribute__((ext_vector_type(2)))  float     f32x2;

union Frag32 { u32x4 u4[2]; v16bf v; };   // 32 bytes = one 16-bit WMMA operand per lane

__device__ __forceinline__ unsigned bf16rne(float x) {
  unsigned u = __float_as_uint(x);
  return (u + 0x7FFFu + ((u >> 16) & 1u)) >> 16;   // round-to-nearest-even
}
__device__ __forceinline__ float lrelu(float x) { return x >= 0.f ? x : NEGS * x; }

// ---------------------------------------------------------------------------
// Pack W (f32, [N,512] row-major) into bf16 WMMA-B fragments.
// Layout: out[((t*16 + k)*32 + lane)*16 + i]  where (this kernel's B order, shared with A):
//   N = t*16 + (lane&15),  K = k*32 + (lane>>4)*16 + i   (contiguous K per half)
// ---------------------------------------------------------------------------
__global__ __launch_bounds__(256) void pack_kernel(const float* __restrict__ W,
                                                   unsigned short* __restrict__ out,
                                                   int ntiles) {
  int idx = blockIdx.x * 256 + threadIdx.x;
  int total = ntiles * 16 * 32 * 16;
  if (idx >= total) return;
  int i    = idx & 15;
  int lane = (idx >> 4) & 31;
  int k    = (idx >> 9) & 15;
  int t    = idx >> 13;
  int n  = t * 16 + (lane & 15);
  int kk = k * 32 + (lane >> 4) * 16 + i;
  out[idx] = (unsigned short)bf16rne(W[(size_t)n * ENCD + kk]);
}

// ---------------------------------------------------------------------------
// query = lrelu(ht @ Wq^T + bq)   [32,128] — tiny
// ---------------------------------------------------------------------------
__global__ __launch_bounds__(128) void query_kernel(const float* __restrict__ ht,
                                                    const float* __restrict__ Wq,
                                                    const float* __restrict__ bq,
                                                    float* __restrict__ qout) {
  int b = blockIdx.x, q = threadIdx.x;
  const f32x4* h = (const f32x4*)(ht + (size_t)b * ENCD);
  const f32x4* w = (const f32x4*)(Wq + (size_t)q * ENCD);
  float acc = bq[q];
  for (int e = 0; e < ENCD / 4; ++e) {
    f32x4 a = h[e], c = w[e];
    acc += a[0]*c[0] + a[1]*c[1] + a[2]*c[2] + a[3]*c[3];
  }
  qout[b * QDIM + q] = lrelu(acc);
}

// ---------------------------------------------------------------------------
// Main fused pass: one 16-row S-tile per workgroup (8 wave32s).
//   key   = lrelu(hs_tile @ Wk^T + bk)      -> energy[16] (masked, stored)
//   feats = lrelu(hs_tile @ Wc^T + bc)      -> c_j = sum_s exp(e_s - m_j)*feats
// 24 N-tiles (8 key + 16 ctx), 3 per wave, K contraction = 16 WMMA steps.
// ---------------------------------------------------------------------------
__global__ __launch_bounds__(256) void fused_kernel(
    const float* __restrict__ hs, const int* __restrict__ state_len,
    const float* __restrict__ bk, const float* __restrict__ bc,
    const float* __restrict__ query,
    const unsigned short* __restrict__ pWk,
    const unsigned short* __restrict__ pWc,
    float* __restrict__ energy_out,   // [32][8192] masked energies (-inf invalid)
    float* __restrict__ m_out,        // [32][512]  per-tile max
    float* __restrict__ cpart)        // [32][512][256] partial contexts
{
  __shared__ __align__(16) unsigned short hsb[16 * ENCD];  // bf16 tile, 16 KB
  __shared__ float q_lds[QDIM];
  __shared__ float bk_lds[QDIM];
  __shared__ float bc_lds[CDIM];
  __shared__ float e_lds[16];
  __shared__ float p_lds[16];

  const int b   = blockIdx.y;
  const int st  = blockIdx.x;
  const int tid = threadIdx.x;
  const int s0  = st * 16;

  // Stage hs tile (16x512 f32) -> LDS bf16, fully coalesced
  const float* hs_tile = hs + ((size_t)b * SLEN + s0) * ENCD;
  for (int i = tid; i < 16 * ENCD / 2; i += 256) {
    f32x2 f = ((const f32x2*)hs_tile)[i];
    ((unsigned*)hsb)[i] = bf16rne(f[0]) | (bf16rne(f[1]) << 16);
  }
  if (tid < QDIM) { q_lds[tid] = query[b * QDIM + tid]; bk_lds[tid] = bk[tid]; }
  if (tid < CDIM) bc_lds[tid] = bc[tid];
  if (tid < 16)   e_lds[tid] = 0.f;
  __syncthreads();

  const int wave = tid >> 5;
  const int lane = tid & 31;
  const int nloc = lane & 15;
  const int half = lane >> 4;

  v8f acc[3] = {};
  const unsigned short* bbase[3];
#pragma unroll
  for (int i = 0; i < 3; ++i) {
    int tile = wave * 3 + i;
    bbase[i] = (tile < 8) ? (pWk + (size_t)tile * (16 * 32 * 16))
                          : (pWc + (size_t)(tile - 8) * (16 * 32 * 16));
  }

  const u32x4* hs4 = (const u32x4*)hsb;
  for (int k = 0; k < 16; ++k) {
    // A fragment: contiguous K per lane half {0..15}/{16..31} (matches packed B)
    Frag32 A;
    int aoff = ((lane & 15) * ENCD + k * 32 + half * 16) >> 3;  // u32x4 units
    A.u4[0] = hs4[aoff];
    A.u4[1] = hs4[aoff + 1];
#pragma unroll
    for (int i = 0; i < 3; ++i) {
      Frag32 Bf;
      const u32x4* bp = (const u32x4*)(bbase[i] + (size_t)(k * 32 + lane) * 16);
      Bf.u4[0] = bp[0];
      Bf.u4[1] = bp[1];
      acc[i] = __builtin_amdgcn_wmma_f32_16x16x32_bf16(
          false, A.v, false, Bf.v, (short)0, acc[i], false, false);
    }
  }

  // ---- key epilogue: energy[s] = sum_n lrelu(key)*query[n] ----
#pragma unroll
  for (int i = 0; i < 3; ++i) {
    int tile = wave * 3 + i;
    if (tile < 8) {
      int ncol = tile * 16 + nloc;
      float qv = q_lds[ncol];
      float bv = bk_lds[ncol];
#pragma unroll
      for (int v = 0; v < 8; ++v) {
        // C/D layout: element v of lane -> row M = v + 8*half, col N = nloc
        float val = lrelu(acc[i][v] + bv) * qv;
        val += __shfl_xor(val, 1);
        val += __shfl_xor(val, 2);
        val += __shfl_xor(val, 4);
        val += __shfl_xor(val, 8);
        if (nloc == 0) atomicAdd(&e_lds[v + 8 * half], val);
      }
    }
  }
  __syncthreads();

  if (tid == 0) {
    int L = state_len[b];
    float m = -INFINITY;
    for (int r = 0; r < 16; ++r)
      if (s0 + r < L) m = fmaxf(m, e_lds[r]);
    for (int r = 0; r < 16; ++r)
      p_lds[r] = (s0 + r < L) ? __expf(e_lds[r] - m) : 0.f;
    m_out[b * STILES + st] = m;
  }
  if (tid < 16) {
    bool valid = (s0 + tid) < state_len[b];
    energy_out[(size_t)b * SLEN + s0 + tid] = valid ? e_lds[tid] : -INFINITY;
  }
  __syncthreads();

  // ---- ctx epilogue: c_j[n] = sum_s p[s] * lrelu(feats[s][n]) ----
#pragma unroll
  for (int i = 0; i < 3; ++i) {
    int tile = wave * 3 + i;
    if (tile >= 8) {
      int ct   = tile - 8;
      int ncol = ct * 16 + nloc;
      float bv = bc_lds[ncol];
      float wsum = 0.f;
#pragma unroll
      for (int v = 0; v < 8; ++v)
        wsum += p_lds[v + 8 * half] * lrelu(acc[i][v] + bv);
      wsum += __shfl_xor(wsum, 16);   // combine rows 0..7 with 8..15
      if (half == 0)
        cpart[((size_t)b * STILES + st) * CDIM + ncol] = wsum;
    }
  }
}

// ---------------------------------------------------------------------------
// Finalize: per batch, global max/Z over masked energies -> alpha; combine
// per-tile partial contexts with exp(m_j - M)/Z.
// ---------------------------------------------------------------------------
__global__ __launch_bounds__(256) void finalize_kernel(
    const float* __restrict__ energy, const float* __restrict__ m_arr,
    const float* __restrict__ cpart,
    float* __restrict__ out_alpha, float* __restrict__ out_ctx) {
  __shared__ float red[256];
  __shared__ float w_lds[STILES];
  __shared__ float Msh, Zsh;
  int b = blockIdx.x, tid = threadIdx.x;
  const float* erow = energy + (size_t)b * SLEN;

  float m = -INFINITY;
  for (int s = tid; s < SLEN; s += 256) m = fmaxf(m, erow[s]);
  red[tid] = m; __syncthreads();
  for (int o = 128; o; o >>= 1) { if (tid < o) red[tid] = fmaxf(red[tid], red[tid + o]); __syncthreads(); }
  if (tid == 0) Msh = red[0];
  __syncthreads();
  float M = Msh;

  float z = 0.f;
  for (int s = tid; s < SLEN; s += 256) z += __expf(erow[s] - M);  // exp(-inf)=0 masks
  red[tid] = z; __syncthreads();
  for (int o = 128; o; o >>= 1) { if (tid < o) red[tid] += red[tid + o]; __syncthreads(); }
  if (tid == 0) Zsh = red[0];
  __syncthreads();
  float invZ = 1.f / Zsh;

  for (int s = tid; s < SLEN; s += 256)
    out_alpha[(size_t)b * SLEN + s] = __expf(erow[s] - M) * invZ;

  for (int j = tid; j < STILES; j += 256)
    w_lds[j] = __expf(m_arr[b * STILES + j] - M);
  __syncthreads();

  // 256 threads = 256 context columns; coalesced over j
  int c = tid;
  const float* cp = cpart + (size_t)b * STILES * CDIM;
  float acc = 0.f;
  for (int j = 0; j < STILES; ++j) acc += w_lds[j] * cp[(size_t)j * CDIM + c];
  out_ctx[b * CDIM + c] = acc * invZ;
}

// ---------------------------------------------------------------------------
extern "C" void kernel_launch(void* const* d_in, const int* in_sizes, int n_in,
                              void* d_out, int out_size, void* d_ws, size_t ws_size,
                              hipStream_t stream) {
  (void)in_sizes; (void)n_in; (void)out_size; (void)ws_size;
  const float* hs = (const float*)d_in[0];
  const float* ht = (const float*)d_in[1];
  const int*   sl = (const int*)d_in[2];
  const float* Wq = (const float*)d_in[3];
  const float* bq = (const float*)d_in[4];
  const float* Wk = (const float*)d_in[5];
  const float* bk = (const float*)d_in[6];
  const float* Wc = (const float*)d_in[7];
  const float* bc = (const float*)d_in[8];
  float* out = (float*)d_out;

  char* ws = (char*)d_ws;
  float*          qbuf  = (float*)(ws + 0);                 //  16 KB
  unsigned short* pWk   = (unsigned short*)(ws + 16384);    // 128 KB
  unsigned short* pWc   = (unsigned short*)(ws + 147456);   // 256 KB
  float*          ebuf  = (float*)(ws + 409600);            //   1 MB
  float*          mbuf  = (float*)(ws + 1458176);           //  64 KB
  float*          cpart = (float*)(ws + 1523712);           //  16 MB

  pack_kernel<<<dim3(65536 / 256), 256, 0, stream>>>(Wk, pWk, QDIM / 16);
  pack_kernel<<<dim3(131072 / 256), 256, 0, stream>>>(Wc, pWc, CDIM / 16);
  query_kernel<<<dim3(BATCH), 128, 0, stream>>>(ht, Wq, bq, qbuf);
  fused_kernel<<<dim3(STILES, BATCH), 256, 0, stream>>>(hs, sl, bk, bc, qbuf,
                                                        pWk, pWc, ebuf, mbuf, cpart);
  finalize_kernel<<<dim3(BATCH), 256, 0, stream>>>(ebuf, mbuf, cpart,
                                                   out, out + (size_t)BATCH * SLEN);
}
